// MultiHeadAttention_89799176225071
// MI455X (gfx1250) — matrix operand order repaired and checked by reading, not compiled, not run on hardware
//
#include <hip/hip_runtime.h>
#include <hip/hip_bf16.h>
#include <math.h>

#define SS 2048
#define DM 1024
#define NH 16
#define DK 64
#define DV 64

typedef unsigned short u16;
typedef __attribute__((ext_vector_type(16))) __bf16 v16bf;
typedef __attribute__((ext_vector_type(8)))  __bf16 v8bf;
typedef __attribute__((ext_vector_type(8)))  float  v8f;

// ---------- helpers ----------

__device__ __forceinline__ u16 f2bf(float f) {
  union { float f; unsigned u; } x; x.f = f;
  unsigned u = x.u;
  unsigned r = u + 0x7FFFu + ((u >> 16) & 1u);   // round-to-nearest-even
  return (u16)(r >> 16);
}

// A-fragment (16x32 bf16, MxK): lane-half hs gets K = hs*16 + e
__device__ __forceinline__ v16bf load_fragA(const u16* p, int hs) {
  union { v16bf v; v8bf h[2]; } u;
  u.h[0] = *reinterpret_cast<const v8bf*>(p + hs * 16);
  u.h[1] = *reinterpret_cast<const v8bf*>(p + hs * 16 + 8);
  return u.v;
}

// B-fragment (32x16 bf16, KxN): lanes 0-15 hold K=0..15, lanes 16-31 hold K=16..31
__device__ __forceinline__ v16bf load_fragB(const u16* p, int hs) {
  const u16* q = p + hs * 16;
  union { v16bf v; v8bf h[2]; } u;
  u.h[0] = *reinterpret_cast<const v8bf*>(q);
  u.h[1] = *reinterpret_cast<const v8bf*>(q + 8);
  return u.v;
}

__device__ __forceinline__ v8f wmma_bf16(v16bf a, v16bf b, v8f c) {
  return __builtin_amdgcn_wmma_f32_16x16x32_bf16(false, a, false, b, (short)0, c,
                                                 false, false);
}

__device__ __forceinline__ float rmax16(float x) {
#pragma unroll
  for (int m = 1; m < 16; m <<= 1) x = fmaxf(x, __shfl_xor(x, m, 32));
  return x;
}
__device__ __forceinline__ float rsum16(float x) {
#pragma unroll
  for (int m = 1; m < 16; m <<= 1) x += __shfl_xor(x, m, 32);
  return x;
}

// ---------- conversion / transpose kernels ----------

__global__ void cvt_f32_bf16_kernel(const float* __restrict__ in, u16* __restrict__ out, int n) {
  int t = blockIdx.x * blockDim.x + threadIdx.x;
  if (t < n) out[t] = f2bf(in[t]);
}

// [NH][DM][DK] f32 -> [NH][DK][DM] bf16 (optionally scaled)
__global__ void tr_w_kernel(const float* __restrict__ in, u16* __restrict__ out, float scale) {
  int t = blockIdx.x * blockDim.x + threadIdx.x;
  if (t >= NH * DM * DK) return;
  int h = t / (DM * DK);
  int rem = t % (DM * DK);
  int din = rem / DK;
  int dout = rem % DK;
  out[((size_t)h * DK + dout) * DM + din] = f2bf(in[t] * scale);
}

// [DM][DM] f32 -> transposed [DM][DM] bf16
__global__ void tr_wo_kernel(const float* __restrict__ in, u16* __restrict__ out) {
  int t = blockIdx.x * blockDim.x + threadIdx.x;
  if (t >= DM * DM) return;
  int din = t / DM, dout = t % DM;
  out[(size_t)dout * DM + din] = f2bf(in[t]);
}

// ---------- unified 32x64-tile GEMM (one wave per tile) ----------
// C[32x64] = A[32xDM] @ B^T[64xDM]  with per-mode epilogue.
// MODE 0: q-proj  -> bf16 row-major [h][SS][DK], bias*0.125 (softmax scale folded)
// MODE 1: k-proj  -> bf16 row-major [h][SS][DK], bias
// MODE 2: v-proj  -> bf16 TRANSPOSED [h][DV][SS], bias
// MODE 3: out-proj-> f32 row-major [SS][DM], bias

template <int MODE>
__global__ void __launch_bounds__(256, 1)
gemm_kernel(const u16* __restrict__ Amat, const u16* __restrict__ Bt,
            const float* __restrict__ bias, void* __restrict__ outp) {
  const int lane = threadIdx.x & 31;
  const int wid  = threadIdx.x >> 5;
  const int task = blockIdx.x * 8 + wid;          // 0..1023
  const int n  = lane & 15;
  const int hs = lane >> 4;

  int h, m0, n0;
  const u16* wbase;
  const float* bb;
  if (MODE == 3) {                                 // task = mtile*16 + nblk
    h = 0;
    m0 = (task >> 4) * 32;
    n0 = (task & 15) * 64;
    wbase = Bt + (size_t)n0 * DM;
    bb = bias + n0;
  } else {                                         // task = h*64 + mtile
    h = task >> 6;
    m0 = (task & 63) * 32;
    n0 = 0;
    wbase = Bt + (size_t)h * DK * DM;
    bb = bias + h * DK;
  }

  v8f acc[2][4];
#pragma unroll
  for (int mt = 0; mt < 2; mt++)
#pragma unroll
    for (int c = 0; c < 4; c++) acc[mt][c] = (v8f)0.f;

  const u16* ar0 = Amat + (size_t)(m0 + n) * DM;
  const u16* ar1 = Amat + (size_t)(m0 + 16 + n) * DM;

  for (int ks = 0; ks < DM; ks += 32) {
    v16bf a0 = load_fragA(ar0 + ks, hs);
    v16bf a1 = load_fragA(ar1 + ks, hs);
    v16bf b0 = load_fragB(wbase + (size_t)(0 * 16 + n) * DM + ks, hs);
    v16bf b1 = load_fragB(wbase + (size_t)(1 * 16 + n) * DM + ks, hs);
    v16bf b2 = load_fragB(wbase + (size_t)(2 * 16 + n) * DM + ks, hs);
    v16bf b3 = load_fragB(wbase + (size_t)(3 * 16 + n) * DM + ks, hs);
    acc[0][0] = wmma_bf16(a0, b0, acc[0][0]);
    acc[1][0] = wmma_bf16(a1, b0, acc[1][0]);
    acc[0][1] = wmma_bf16(a0, b1, acc[0][1]);
    acc[1][1] = wmma_bf16(a1, b1, acc[1][1]);
    acc[0][2] = wmma_bf16(a0, b2, acc[0][2]);
    acc[1][2] = wmma_bf16(a1, b2, acc[1][2]);
    acc[0][3] = wmma_bf16(a0, b3, acc[0][3]);
    acc[1][3] = wmma_bf16(a1, b3, acc[1][3]);
  }

  // epilogue
#pragma unroll
  for (int c = 0; c < 4; c++) {
    int col = c * 16 + n;
    float bv = bb[col] * ((MODE == 0) ? 0.125f : 1.0f);
#pragma unroll
    for (int mt = 0; mt < 2; mt++) {
#pragma unroll
      for (int r = 0; r < 8; r++) {
        int row = m0 + mt * 16 + r + 8 * hs;
        float val = acc[mt][c][r] + bv;
        if (MODE == 0 || MODE == 1) {
          u16* oh = (u16*)outp + (size_t)h * SS * DK;
          oh[(size_t)row * DK + col] = f2bf(val);
        } else if (MODE == 2) {
          u16* oh = (u16*)outp + (size_t)h * DV * SS;
          oh[(size_t)col * SS + row] = f2bf(val);       // V stored transposed
        } else {
          ((float*)outp)[(size_t)row * DM + n0 + col] = val;
        }
      }
    }
  }
}

// ---------- flash attention: one wave per (head, 16-query tile) ----------

__global__ void __launch_bounds__(256, 1)
attn_kernel(const u16* __restrict__ qb, const u16* __restrict__ kb,
            const u16* __restrict__ vbT, u16* __restrict__ cc) {
  __shared__ __align__(16) u16 Plds[8][16 * 32];
  const int lane = threadIdx.x & 31;
  const int wid  = threadIdx.x >> 5;
  const int task = blockIdx.x * 8 + wid;
  const int h  = task >> 7;
  const int m0 = (task & 127) << 4;
  const int n  = lane & 15;
  const int hs = lane >> 4;

  const u16* qrow = qb + (size_t)h * SS * DK + (size_t)(m0 + n) * DK;
  const v16bf aq0 = load_fragA(qrow, hs);        // K 0..31
  const v16bf aq1 = load_fragA(qrow + 32, hs);   // K 32..63

  const u16* kbase = kb + (size_t)h * SS * DK;
  const u16* vbase = vbT + (size_t)h * DV * SS;

  v8f o[4];
#pragma unroll
  for (int c = 0; c < 4; c++) o[c] = (v8f)0.f;
  float rmax[8], rsum[8];
#pragma unroll
  for (int r = 0; r < 8; r++) { rmax[r] = -INFINITY; rsum[r] = 0.f; }

  u16* pl = &Plds[wid][0];

  for (int j0 = 0; j0 < SS; j0 += 32) {
    // ---- scores: 16x32 tile of q·kT (q pre-scaled by 1/8) ----
    const u16* krow0 = kbase + (size_t)(j0 + n) * DK;
    const u16* krow1 = kbase + (size_t)(j0 + 16 + n) * DK;
    v16bf bk00 = load_fragB(krow0, hs);
    v16bf bk01 = load_fragB(krow0 + 32, hs);
    v16bf bk10 = load_fragB(krow1, hs);
    v16bf bk11 = load_fragB(krow1 + 32, hs);

    // V fragments for this key block: independent of P; softmax VALU hides them
    v16bf bv0 = load_fragB(vbase + (size_t)(0 * 16 + n) * SS + j0, hs);
    v16bf bv1 = load_fragB(vbase + (size_t)(1 * 16 + n) * SS + j0, hs);
    v16bf bv2 = load_fragB(vbase + (size_t)(2 * 16 + n) * SS + j0, hs);
    v16bf bv3 = load_fragB(vbase + (size_t)(3 * 16 + n) * SS + j0, hs);

    v8f s0 = (v8f)0.f, s1 = (v8f)0.f;
    s0 = wmma_bf16(aq0, bk00, s0);
    s0 = wmma_bf16(aq1, bk01, s0);
    s1 = wmma_bf16(aq0, bk10, s1);
    s1 = wmma_bf16(aq1, bk11, s1);

    // warm L2/WGP$ for next key block (global_prefetch_b8)
    if (j0 + 32 < SS) {
      __builtin_prefetch(kbase + (size_t)(j0 + 32 + n) * DK, 0, 3);
      __builtin_prefetch(kbase + (size_t)(j0 + 48 + n) * DK, 0, 3);
    }

    // ---- online softmax per row ----
#pragma unroll
    for (int r = 0; r < 8; r++) {
      float mt = rmax16(fmaxf(s0[r], s1[r]));
      float mn = fmaxf(rmax[r], mt);
      float corr = __expf(rmax[r] - mn);
      rmax[r] = mn;
      float p0 = __expf(s0[r] - mn);
      float p1 = __expf(s1[r] - mn);
      rsum[r] = rsum[r] * corr + rsum16(p0 + p1);
      o[0][r] *= corr; o[1][r] *= corr; o[2][r] *= corr; o[3][r] *= corr;
      int row = r + 8 * hs;
      pl[row * 32 + n]      = f2bf(p0);
      pl[row * 32 + 16 + n] = f2bf(p1);
    }

    // re-swizzle P (D-layout) -> A-fragment via LDS (same-wave DS ops are in order)
    v16bf pa = load_fragA(pl + n * 32, hs);

    // ---- P(16x32) @ V(32x64): operands already resident ----
    o[0] = wmma_bf16(pa, bv0, o[0]);
    o[1] = wmma_bf16(pa, bv1, o[1]);
    o[2] = wmma_bf16(pa, bv2, o[2]);
    o[3] = wmma_bf16(pa, bv3, o[3]);
  }

  // finalize and write concat [s][h*64+dv] as bf16 (fast v_rcp_f32 for 1/rowsum)
#pragma unroll
  for (int r = 0; r < 8; r++) {
    float inv = __builtin_amdgcn_rcpf(rsum[r]);
    int row = m0 + r + 8 * hs;
#pragma unroll
    for (int c = 0; c < 4; c++) {
      cc[(size_t)row * (NH * DV) + h * DV + c * 16 + n] = f2bf(o[c][r] * inv);
    }
  }
}

// ---------- launch ----------

extern "C" void kernel_launch(void* const* d_in, const int* in_sizes, int n_in,
                              void* d_out, int out_size, void* d_ws, size_t ws_size,
                              hipStream_t stream) {
  const float* emb = (const float*)d_in[0];
  const float* Wq  = (const float*)d_in[1];
  const float* bq  = (const float*)d_in[2];
  const float* Wk  = (const float*)d_in[3];
  const float* bk  = (const float*)d_in[4];
  const float* Wv  = (const float*)d_in[5];
  const float* bv  = (const float*)d_in[6];
  const float* Wo  = (const float*)d_in[7];
  const float* bo  = (const float*)d_in[8];
  float* out = (float*)d_out;

  // workspace carve-up (bf16 elements)
  u16* ws = (u16*)d_ws;
  u16* emb_bf = ws;                     size_t off = (size_t)SS * DM;
  u16* WqT = ws + off;                  off += (size_t)NH * DK * DM;
  u16* WkT = ws + off;                  off += (size_t)NH * DK * DM;
  u16* WvT = ws + off;                  off += (size_t)NH * DK * DM;
  u16* WoT = ws + off;                  off += (size_t)DM * DM;
  u16* qb  = ws + off;                  off += (size_t)NH * SS * DK;
  u16* kb  = ws + off;                  off += (size_t)NH * SS * DK;
  u16* vbT = ws + off;                  off += (size_t)NH * DV * SS;
  u16* cc  = ws + off;                  off += (size_t)SS * NH * DV;

  const int BT = 256;
  // conversions
  cvt_f32_bf16_kernel<<<(SS * DM + BT - 1) / BT, BT, 0, stream>>>(emb, emb_bf, SS * DM);
  tr_w_kernel<<<(NH * DM * DK + BT - 1) / BT, BT, 0, stream>>>(Wq, WqT, 0.125f); // fold 1/sqrt(dk)
  tr_w_kernel<<<(NH * DM * DK + BT - 1) / BT, BT, 0, stream>>>(Wk, WkT, 1.0f);
  tr_w_kernel<<<(NH * DM * DK + BT - 1) / BT, BT, 0, stream>>>(Wv, WvT, 1.0f);
  tr_wo_kernel<<<(DM * DM + BT - 1) / BT, BT, 0, stream>>>(Wo, WoT);

  // GEMMs: 1024 wave-tasks each (32x64 tile per wave), 8 waves per block
  gemm_kernel<0><<<128, BT, 0, stream>>>(emb_bf, WqT, bq, qb);
  gemm_kernel<1><<<128, BT, 0, stream>>>(emb_bf, WkT, bk, kb);
  gemm_kernel<2><<<128, BT, 0, stream>>>(emb_bf, WvT, bv, vbT);

  // flash attention: 2048 wave-tasks
  attn_kernel<<<256, BT, 0, stream>>>(qb, kb, vbT, cc);

  // output projection
  gemm_kernel<3><<<128, BT, 0, stream>>>(cc, WoT, bo, out);
}
